// ScanBlock_31439160606834
// MI455X (gfx1250) — matrix operand order repaired and checked by reading, not compiled, not run on hardware
//
#include <hip/hip_runtime.h>

typedef __attribute__((ext_vector_type(16))) __bf16 v16bf;
typedef __attribute__((ext_vector_type(8)))  __bf16 v8bf;
typedef __attribute__((ext_vector_type(8)))  float  v8f;

#define BM 128
#define BN 64
#define KT 32
#define LDST 40   // KT + 8 bf16 padding: 80B row stride, multiple of 16B

// f32 -> bf16: use the native conversion (gfx1250 has bf16 ALU/cvt hardware);
// clang lowers fptrunc-to-bfloat to v_cvt (packs pairs where possible).
__device__ __forceinline__ __bf16 f2bf(float f) {
    return (__bf16)f;
}

// flip permutation (involution): rows [start, L) reversed within each batch
__device__ __forceinline__ int perm_row(int m, int L, int start) {
    int b = m / L, l = m % L;
    int pl = (l < start) ? l : (L - 1 + start - l);
    return b * L + pl;
}

// out[m][n] = sum_k A[rowmap(m)][k] * W[n][k]  (+bias[n]) (+residual[m][n]) (act)
// A: M x K row-major (lda), W: N x K row-major (ldw), out: M x N (ldo)
// PERM: gather A rows through the flip permutation. KTAIL: K % KT != 0.
template<bool PERM, bool KTAIL>
__global__ __launch_bounds__(256)
void gemm_bf16_wmma_kernel(const float* __restrict__ A, int lda,
                           const float* __restrict__ W, int ldw,
                           const float* __restrict__ bias,
                           const float* __restrict__ residual,
                           float* __restrict__ out, int ldo,
                           int M, int N, int K,
                           int L, const int* __restrict__ startp,
                           int act /*0=none, 1=softplus*/) {
    __shared__ __align__(16) __bf16 lAs[BM * LDST];
    __shared__ __align__(16) __bf16 lBs[BN * LDST];

    const int tid   = threadIdx.x;
    const int lane  = tid & 31;
    const int wave  = tid >> 5;        // 0..7
    const int waveM = wave >> 1;       // 0..3 -> 32-row slices
    const int waveN = wave & 1;        // 0..1 -> 32-col slices
    const int m0 = blockIdx.x * BM;
    const int n0 = blockIdx.y * BN;

    // ---- hoist all row addressing out of the K loop ----
    // A staging: thread handles rows rA[it] = (tid>>3) + it*32, cols (tid&7)*4..+3
    const int rA = tid >> 3;
    const int cA = (tid & 7) * 4;
    const float* arow[4];
    {
        int start = 0;
        if (PERM) start = *startp;
        #pragma unroll
        for (int it = 0; it < 4; ++it) {
            int gm = m0 + rA + it * 32;
            int gr = PERM ? perm_row(gm, L, start) : gm;
            arow[it] = A + (long)gr * lda + cA;
        }
    }
    // B staging: thread handles rows rA + it*32 of the weight tile (only it=0,1)
    const float* brow[2];
    bool bval[2];
    #pragma unroll
    for (int it = 0; it < 2; ++it) {
        int gn = n0 + rA + it * 32;
        bval[it] = (gn < N);
        brow[it] = W + (long)(bval[it] ? gn : 0) * ldw + cA;
    }
    __bf16* dA0 = &lAs[rA * LDST + cA];
    __bf16* dB0 = &lBs[rA * LDST + cA];

    v8f acc[2][2];
    #pragma unroll
    for (int mi = 0; mi < 2; ++mi)
        #pragma unroll
        for (int ni = 0; ni < 2; ++ni)
            acc[mi][ni] = (v8f){0.f,0.f,0.f,0.f,0.f,0.f,0.f,0.f};

    const int fr  = lane & 15;         // fragment row (M for A, N for B)
    const int kha = (lane >> 4) * 8;   // A K-half base (ISA 16-bit A 16x32 layout)
    const int khb = (lane >> 4) * 8;   // B K-half base (same k order as A)
    const __bf16* pa0 = &lAs[(waveM * 32 + fr) * LDST + kha];
    const __bf16* pb0 = &lBs[(waveN * 32 + fr) * LDST + khb];

    float4 ra[4], rb[2];

    // register double buffer: global loads for K-step k0
    auto load_regs = [&](int k0) {
        #pragma unroll
        for (int it = 0; it < 4; ++it) {
            if (KTAIL) {
                float4 v;
                const float* ap = arow[it] + k0;
                v.x = (k0 + cA + 0 < K) ? ap[0] : 0.f;
                v.y = (k0 + cA + 1 < K) ? ap[1] : 0.f;
                v.z = (k0 + cA + 2 < K) ? ap[2] : 0.f;
                v.w = (k0 + cA + 3 < K) ? ap[3] : 0.f;
                ra[it] = v;
            } else {
                ra[it] = *(const float4*)(arow[it] + k0);
            }
        }
        #pragma unroll
        for (int it = 0; it < 2; ++it) {
            float4 v = {0.f, 0.f, 0.f, 0.f};
            if (bval[it]) {
                const float* wp = brow[it] + k0;
                if (KTAIL) {
                    v.x = (k0 + cA + 0 < K) ? wp[0] : 0.f;
                    v.y = (k0 + cA + 1 < K) ? wp[1] : 0.f;
                    v.z = (k0 + cA + 2 < K) ? wp[2] : 0.f;
                    v.w = (k0 + cA + 3 < K) ? wp[3] : 0.f;
                } else {
                    v = *(const float4*)wp;
                }
            }
            rb[it] = v;
        }
    };

    load_regs(0);

    for (int k0 = 0; k0 < K; k0 += KT) {
        // commit staged registers to LDS (previous iteration's trailing barrier
        // guarantees all waves are done reading the old tiles)
        #pragma unroll
        for (int it = 0; it < 4; ++it) {
            __bf16* d = dA0 + it * 32 * LDST;
            d[0] = f2bf(ra[it].x); d[1] = f2bf(ra[it].y);
            d[2] = f2bf(ra[it].z); d[3] = f2bf(ra[it].w);
        }
        #pragma unroll
        for (int it = 0; it < 2; ++it) {
            __bf16* d = dB0 + it * 32 * LDST;
            d[0] = f2bf(rb[it].x); d[1] = f2bf(rb[it].y);
            d[2] = f2bf(rb[it].z); d[3] = f2bf(rb[it].w);
        }
        __syncthreads();

        // issue next K-step's global loads while WMMAs consume current LDS tiles
        if (k0 + KT < K) load_regs(k0 + KT);

        #pragma unroll
        for (int mi = 0; mi < 2; ++mi) {
            const __bf16* pa = pa0 + mi * 16 * LDST;
            v8bf alo = *(const v8bf*)pa;
            v8bf ahi = *(const v8bf*)(pa + 16);
            v16bf afrag = __builtin_shufflevector(alo, ahi,
                0,1,2,3,4,5,6,7,8,9,10,11,12,13,14,15);
            #pragma unroll
            for (int ni = 0; ni < 2; ++ni) {
                const __bf16* pb = pb0 + ni * 16 * LDST;
                v8bf blo = *(const v8bf*)pb;
                v8bf bhi = *(const v8bf*)(pb + 16);
                v16bf bfrag = __builtin_shufflevector(blo, bhi,
                    0,1,2,3,4,5,6,7,8,9,10,11,12,13,14,15);
                acc[mi][ni] = __builtin_amdgcn_wmma_f32_16x16x32_bf16(
                    false, afrag, false, bfrag, (short)0, acc[mi][ni], false, false);
            }
        }
        __syncthreads();
    }

    // ---- epilogue: C/D layout n = lane&15, m = (lane>>4)*8 + r ----
    const int rh = (lane >> 4) * 8;
    #pragma unroll
    for (int mi = 0; mi < 2; ++mi) {
        #pragma unroll
        for (int ni = 0; ni < 2; ++ni) {
            int gn = n0 + waveN * 32 + ni * 16 + fr;
            if (gn >= N) continue;
            float bv = bias ? bias[gn] : 0.f;
            #pragma unroll
            for (int r = 0; r < 8; ++r) {
                int gm = m0 + waveM * 32 + mi * 16 + rh + r;
                float val = acc[mi][ni][r] + bv;
                if (residual) val += residual[(long)gm * ldo + gn];
                if (act == 1) val = (val > 20.f) ? val : log1pf(__expf(val));
                out[(long)gm * ldo + gn] = val;
            }
        }
    }
}

__global__ __launch_bounds__(256)
void rmsnorm_kernel(const float* __restrict__ x, const float* __restrict__ w,
                    float* __restrict__ y, int D) {
    __shared__ float red[256];
    long row = blockIdx.x;
    const float* xr = x + row * D;
    float s = 0.f;
    for (int j = threadIdx.x; j < D; j += 256) { float v = xr[j]; s += v * v; }
    red[threadIdx.x] = s;
    __syncthreads();
    for (int st = 128; st > 0; st >>= 1) {
        if (threadIdx.x < st) red[threadIdx.x] += red[threadIdx.x + st];
        __syncthreads();
    }
    float scale = rsqrtf(red[0] / (float)D + 1e-5f);
    for (int j = threadIdx.x; j < D; j += 256)
        y[row * D + j] = xr[j] * scale * w[j];
}

__global__ __launch_bounds__(256)
void conv_silu_kernel(const float* __restrict__ xz, const float* __restrict__ cw,
                      const float* __restrict__ cb, float* __restrict__ u,
                      int L, int Di, int ldxz, long total) {
    long idx = (long)blockIdx.x * blockDim.x + threadIdx.x;
    if (idx >= total) return;
    int  c  = (int)(idx % Di);
    long bl = idx / Di;
    int  l  = (int)(bl % L);
    long b  = bl / L;
    float acc = cb[c];
    #pragma unroll
    for (int k = 0; k < 4; ++k) {
        int ll = l - 3 + k;
        if (ll >= 0) acc += cw[k * Di + c] * xz[(b * L + ll) * (long)ldxz + c];
    }
    u[idx] = acc / (1.f + __expf(-acc));   // silu
}

#define SWZ_ADD(v, imm) \
    (v) += __builtin_bit_cast(float, __builtin_amdgcn_ds_swizzle(__builtin_bit_cast(int, (v)), (imm)))

// Selective scan: 49152 independent (b,d,n) recurrences.
// lane = state index n within a 16-lane group; 2 channels per wave, 16 per block.
__global__ __launch_bounds__(256)
void scan_kernel(const float* __restrict__ delta, const float* __restrict__ u,
                 const float* __restrict__ xdbc, int ldx,
                 const float* __restrict__ z, int ldz,
                 const float* __restrict__ A_log, const float* __restrict__ Dp,
                 float* __restrict__ ys, int L, int Di, int R) {
    int g   = threadIdx.x >> 4;
    int n   = threadIdx.x & 15;
    int cid = blockIdx.x * 16 + g;
    int b = cid / Di, d = cid % Di;

    float A    = -__expf(A_log[d * 16 + n]);
    float Dpar = Dp[d];
    float h = 0.f;
    long rowBL = (long)b * L;
    for (int l = 0; l < L; ++l) {
        long base = rowBL + l;
        float dv = delta[base * Di + d];
        float uv = u[base * Di + d];
        float Bv = xdbc[base * ldx + R + n];
        float Cv = xdbc[base * ldx + R + 16 + n];
        float dA = __expf(dv * A);
        h = dA * h + (dv * uv) * Bv;
        float yp = h * Cv;
        // butterfly sum over the 16 state lanes (xor 1,2,4,8 stays in-group)
        SWZ_ADD(yp, 0x041F);
        SWZ_ADD(yp, 0x081F);
        SWZ_ADD(yp, 0x101F);
        SWZ_ADD(yp, 0x201F);
        float zv = z[base * ldz + d];
        float y  = yp + uv * Dpar;
        y *= zv / (1.f + __expf(-zv));     // * silu(z)
        if (n == 0) ys[base * Di + d] = y;
    }
}

extern "C" void kernel_launch(void* const* d_in, const int* in_sizes, int n_in,
                              void* d_out, int out_size, void* d_ws, size_t ws_size,
                              hipStream_t stream) {
    const float* x        = (const float*)d_in[0];
    const float* in_w     = (const float*)d_in[1];
    const float* in_b     = (const float*)d_in[2];
    const float* norm_w   = (const float*)d_in[3];
    const float* W_in     = (const float*)d_in[4];
    const float* conv_w   = (const float*)d_in[5];
    const float* conv_b   = (const float*)d_in[6];
    const float* x_proj_w = (const float*)d_in[7];
    const float* dt_w     = (const float*)d_in[8];
    const float* dt_b     = (const float*)d_in[9];
    const float* A_log    = (const float*)d_in[10];
    const float* D_param  = (const float*)d_in[11];
    const float* out_w    = (const float*)d_in[12];
    const float* out_b    = (const float*)d_in[13];
    const float* outp_w   = (const float*)d_in[14];
    const float* outp_b   = (const float*)d_in[15];
    const int*   startp   = (const int*)d_in[16];

    const int Bb = 2, L = 4096, Din = 1024, D = 768, Di = 1536, R = 48;
    const int M = Bb * L;                       // 8192 rows
    const int LDXZ = 2 * Di;                    // 3072
    const int LDXDBC = R + 32;                  // 80

    float* ws    = (float*)d_ws;
    float* hf    = ws; ws += (long)M * D;       // flipped h (residual later)
    float* nb    = ws; ws += (long)M * D;       // rmsnorm output
    float* xz    = ws; ws += (long)M * LDXZ;    // [u0 | z]
    float* ubuf  = ws; ws += (long)M * Di;      // conv+silu output
    float* xdbc  = ws; ws += (long)M * LDXDBC;  // [dt | B | C]
    float* dlt   = ws; ws += (long)M * Di;      // softplus delta
    float* ys    = ws; ws += (long)M * Di;      // scan output (gated)
    float* o1    = ws; ws += (long)M * D;       // pre-flip-back output

    dim3 blk(256);

    // 1) hf = x[perm] @ in_proj_w^T + in_proj_b   (flip fused as row gather)
    gemm_bf16_wmma_kernel<true, false><<<dim3(M / BM, D / BN), blk, 0, stream>>>(
        x, Din, in_w, Din, in_b, nullptr, hf, D, M, D, Din, L, startp, 0);

    // 2) RMSNorm over D
    rmsnorm_kernel<<<M, blk, 0, stream>>>(hf, norm_w, nb, D);

    // 3) xz = nb @ W_in^T
    gemm_bf16_wmma_kernel<false, false><<<dim3(M / BM, LDXZ / BN), blk, 0, stream>>>(
        nb, D, W_in, D, nullptr, nullptr, xz, LDXZ, M, LDXZ, D, L, startp, 0);

    // 4) depthwise causal conv(K=4) + bias + silu on xz[:, :Di] -> ubuf
    long totc = (long)M * Di;
    conv_silu_kernel<<<(unsigned)((totc + 255) / 256), blk, 0, stream>>>(
        xz, conv_w, conv_b, ubuf, L, Di, LDXZ, totc);

    // 5) xdbc = ubuf @ x_proj_w^T  (N=80, edge-guarded)
    gemm_bf16_wmma_kernel<false, false><<<dim3(M / BM, (LDXDBC + BN - 1) / BN), blk, 0, stream>>>(
        ubuf, Di, x_proj_w, Di, nullptr, nullptr, xdbc, LDXDBC, M, LDXDBC, Di, L, startp, 0);

    // 6) delta = softplus(xdbc[:, :R] @ dt_proj_w^T + dt_proj_b)  (K=48 -> tail path)
    gemm_bf16_wmma_kernel<false, true><<<dim3(M / BM, Di / BN), blk, 0, stream>>>(
        xdbc, LDXDBC, dt_w, R, dt_b, nullptr, dlt, Di, M, Di, R, L, startp, 1);

    // 7) selective scan + skip (u*D) + silu(z) gating -> ys
    scan_kernel<<<(Bb * Di) / 16, blk, 0, stream>>>(
        dlt, ubuf, xdbc, LDXDBC, xz + Di, LDXZ, A_log, D_param, ys, L, Di, R);

    // 8) o1 = ys @ out_w^T + out_b + hf   (residual in flipped row space)
    gemm_bf16_wmma_kernel<false, false><<<dim3(M / BM, D / BN), blk, 0, stream>>>(
        ys, Di, out_w, Di, out_b, hf, o1, D, M, D, Di, L, startp, 0);

    // 9) out = o1[perm] @ outp_w^T + outp_b  (flip-back fused: perm is an involution)
    gemm_bf16_wmma_kernel<true, false><<<dim3(M / BM, Din / BN), blk, 0, stream>>>(
        o1, D, outp_w, D, outp_b, nullptr, (float*)d_out, Din, M, Din, D, L, startp, 0);
}
